// inter_feat_linear_14499809591884
// MI455X (gfx1250) — matrix operand order repaired and checked by reading, not compiled, not run on hardware
//
#include <hip/hip_runtime.h>
#include <hip/hip_bf16.h>

// ---- CDNA5 WMMA types -------------------------------------------------------
typedef __attribute__((ext_vector_type(16))) __bf16       v16bf;
typedef __attribute__((ext_vector_type(8)))  float        v8f;
typedef __attribute__((ext_vector_type(4)))  unsigned int v4u;
typedef __attribute__((ext_vector_type(8)))  unsigned int v8u;

// ---- problem constants ------------------------------------------------------
constexpr int kR    = 512;
constexpr int kC    = 512;
constexpr int kRC   = kR * kC;        // 262144 grid pixels
constexpr int kRIn  = 256;
constexpr int kCIn  = 256;
constexpr int kM    = kRIn * kCIn;    // 65536 query points
constexpr int kD    = 64;             // feature dim
constexpr int kCls  = 21;             // classes
constexpr int kTap  = 25;             // 5x5 neighborhood

__device__ __forceinline__ unsigned short f2bf(float f) {
    unsigned int u = __float_as_uint(f);
    u += 0x7FFFu + ((u >> 16) & 1u);          // round-to-nearest-even
    return (unsigned short)(u >> 16);
}

// ---------------------------------------------------------------------------
// Kernel 1: bilinear upsample feat (64,256,256) -> channels-last bf16
// (512*512, 64) + per-pixel squared norm (f32). One thread per output pixel.
// ---------------------------------------------------------------------------
__global__ __launch_bounds__(256) void upsample_kernel(
    const float* __restrict__ feat,
    unsigned short* __restrict__ fg,
    float* __restrict__ norms)
{
    int p = blockIdx.x * 256 + threadIdx.x;
    int y = p / kC, x = p % kC;
    // half-pixel-center bilinear, scale 2x
    float sy = 0.5f * (float)y - 0.25f;
    float sx = 0.5f * (float)x - 0.25f;
    int iy0 = (int)floorf(sy), ix0 = (int)floorf(sx);
    float fy = sy - (float)iy0, fx = sx - (float)ix0;
    int y0 = max(iy0, 0), y1 = min(iy0 + 1, kRIn - 1);
    int x0 = max(ix0, 0), x1 = min(ix0 + 1, kCIn - 1);
    float w00 = (1.f - fy) * (1.f - fx), w01 = (1.f - fy) * fx;
    float w10 = fy * (1.f - fx),          w11 = fy * fx;
    int b00 = y0 * kCIn + x0, b01 = y0 * kCIn + x1;
    int b10 = y1 * kCIn + x0, b11 = y1 * kCIn + x1;

    unsigned int* dst = (unsigned int*)(fg + (size_t)p * kD);
    float nrm = 0.f;
#pragma unroll 4
    for (int d = 0; d < kD; d += 2) {
        const float* f0 = feat + (size_t)d       * kM;
        const float* f1 = feat + (size_t)(d + 1) * kM;
        float a = w00 * f0[b00] + w01 * f0[b01] + w10 * f0[b10] + w11 * f0[b11];
        float b = w00 * f1[b00] + w01 * f1[b01] + w10 * f1[b10] + w11 * f1[b11];
        nrm += a * a + b * b;
        dst[d >> 1] = (unsigned int)f2bf(a) | ((unsigned int)f2bf(b) << 16);
    }
    norms[p] = nrm;
}

// ---------------------------------------------------------------------------
// Kernel 2: query features -> channels-last bf16 (65536,64) + query norms.
// ---------------------------------------------------------------------------
__global__ __launch_bounds__(256) void query_kernel(
    const float* __restrict__ feat,
    unsigned short* __restrict__ qf,
    float* __restrict__ qn)
{
    int m = blockIdx.x * 256 + threadIdx.x;
    unsigned int* dst = (unsigned int*)(qf + (size_t)m * kD);
    float nrm = 0.f;
#pragma unroll 4
    for (int d = 0; d < kD; d += 2) {
        float a = feat[(size_t)d       * kM + m];
        float b = feat[(size_t)(d + 1) * kM + m];
        nrm += a * a + b * b;
        dst[d >> 1] = (unsigned int)f2bf(a) | ((unsigned int)f2bf(b) << 16);
    }
    qn[m] = nrm;
}

// ---------------------------------------------------------------------------
// Kernel 3: labels (21,512,512) -> channels-last (512*512, 21) so one tap's
// 21 class values are a single contiguous ~84B read.
// ---------------------------------------------------------------------------
__global__ __launch_bounds__(256) void lbst_kernel(
    const float* __restrict__ lbs,
    float* __restrict__ lbsT)
{
    int p = blockIdx.x * 256 + threadIdx.x;
#pragma unroll
    for (int c = 0; c < kCls; ++c)
        lbsT[(size_t)p * kCls + c] = lbs[(size_t)c * kRC + p];
}

// ---------------------------------------------------------------------------
// Kernel 4: main. One wave32 per query point; 8 waves / block.
// Dots via V_WMMA_F32_16X16X32_BF16 (A = 16 taps x 32K, B = query feature
// broadcast across 16 columns), dist via norm expansion + exp, voting via
// channels-last label gather + wave shfl_xor reductions.
// ---------------------------------------------------------------------------
__global__ __launch_bounds__(256) void vote_kernel(
    const float* __restrict__ grid_inte, const float* __restrict__ grid,
    const float* __restrict__ wptr,
    const unsigned short* __restrict__ fg, const float* __restrict__ norms,
    const unsigned short* __restrict__ qf, const float* __restrict__ qn,
    const float* __restrict__ lbsT, float* __restrict__ out)
{
    __shared__ float s_dot[8][32];
    __shared__ float s_out[8][24];

    const int lane = threadIdx.x & 31;
    const int wv   = threadIdx.x >> 5;
    const int m    = blockIdx.x * 8 + wv;

    // grid spacing (== 1.0 for meshgrid, but stay faithful to reference)
    const float dxg = grid[(2 * kC + 2) * 2 + 0] - grid[(1 * kC + 2) * 2 + 0];
    const float dyg = grid[(2 * kC + 2) * 2 + 1] - grid[(2 * kC + 1) * 2 + 1];
    const float wei = wptr[0];

    const int rp0 = (int)floorf(grid_inte[2 * m]     / dxg);
    const int rp1 = (int)floorf(grid_inte[2 * m + 1] / dyg);
    const float nm = qn[m];

    // per-lane voting tap (lane t handles tap t; lanes >= 25 clamp to tap 24,
    // masked later via dist = 0)
    int t   = min(lane, kTap - 1);
    int vr  = min(max(rp0 + t / 5 - 2, 0), kR - 1);
    int vc  = min(max(rp1 + t % 5 - 2, 0), kC - 1);
    int vpix = vr * kC + vc;

    // ---- WMMA dot products: dot_t = fg[idx_t] . f_m  -----------------------
    const int row16 = lane & 15;   // A-matrix row (tap within group)
    const int khalf = lane >> 4;   // which K-half this lane carries
    const unsigned short* qbase = qf + (size_t)m * kD;

#pragma unroll
    for (int g = 0; g < 2; ++g) {
        int tt = min(g * 16 + row16, kTap - 1);
        int ar = min(max(rp0 + tt / 5 - 2, 0), kR - 1);
        int ac = min(max(rp1 + tt % 5 - 2, 0), kC - 1);
        const unsigned short* abase = fg + (size_t)(ar * kC + ac) * kD;

        v8f acc = {0.f, 0.f, 0.f, 0.f, 0.f, 0.f, 0.f, 0.f};
#pragma unroll
        for (int k0 = 0; k0 < kD; k0 += 32) {
            // A 16x32 bf16 layout: lanes 0-15 hold K {0-7,16-23}, lanes 16-31
            // hold K {8-15,24-31} of their row.
            v8u au, bu;
            au.lo = *(const v4u*)(abase + k0 + khalf * 8);
            au.hi = *(const v4u*)(abase + k0 + 16 + khalf * 8);
            // B 32x16 bf16 broadcast: lanes 0-15 hold K {0-7,16-23}, lanes 16-31
            // hold K {8-15,24-31}; every column = f_m.
            bu.lo = *(const v4u*)(qbase + k0 + khalf * 8);
            bu.hi = *(const v4u*)(qbase + k0 + 16 + khalf * 8);
            v16bf A = __builtin_bit_cast(v16bf, au);
            v16bf B = __builtin_bit_cast(v16bf, bu);
            acc = __builtin_amdgcn_wmma_f32_16x16x32_bf16(
                false, A, false, B, (short)0, acc, false, false);
        }
        // All C columns equal; lane0 holds rows 0-7 (VGPR r = M=r), lane16
        // holds rows 8-15 (VGPR r = M=8+r).
        if ((lane & 15) == 0) {
            int base = g * 16 + (khalf ? 8 : 0);
#pragma unroll
            for (int r = 0; r < 8; ++r) s_dot[wv][base + r] = acc[r];
        }
    }
    __syncthreads();

    // ---- dist_t = exp(-wei * (||fg||^2 - 2 dot + ||f||^2)/64) --------------
    float dist = 0.f;
    if (lane < kTap) {
        float dot = s_dot[wv][lane];
        float nt  = norms[vpix];
        float msq = (nt - 2.f * dot + nm) * (1.0f / (float)kD);
        dist = __expf(-wei * msq);
    }
    float dsum = dist;
#pragma unroll
    for (int s = 16; s; s >>= 1) dsum += __shfl_xor(dsum, s, 32);
    float inv = 1.0f / fmaxf(dsum, 1e-15f);

    // ---- voting: one contiguous 21-float label row per tap -----------------
    const float* lrow = lbsT + (size_t)vpix * kCls;
    float vote[kCls];
#pragma unroll
    for (int c = 0; c < kCls; ++c) vote[c] = lrow[c] * dist;
#pragma unroll
    for (int c = 0; c < kCls; ++c) {
        float v = vote[c];
#pragma unroll
        for (int s = 16; s; s >>= 1) v += __shfl_xor(v, s, 32);
        if (lane == 0) s_out[wv][c] = v * inv;
    }
    __syncthreads();

    // coalesced block-wide output write: out[c, m]
    int j = threadIdx.x;
    if (j < kCls * 8) {
        int c = j >> 3, w = j & 7;
        out[(size_t)c * kM + blockIdx.x * 8 + w] = s_out[w][c];
    }
}

// ---------------------------------------------------------------------------
extern "C" void kernel_launch(void* const* d_in, const int* in_sizes, int n_in,
                              void* d_out, int out_size, void* d_ws, size_t ws_size,
                              hipStream_t stream) {
    (void)in_sizes; (void)n_in; (void)out_size; (void)ws_size;
    const float* grid_inte = (const float*)d_in[0];
    const float* grid      = (const float*)d_in[1];
    const float* feat      = (const float*)d_in[2];
    const float* lbs       = (const float*)d_in[3];
    const float* wei       = (const float*)d_in[4];
    float* out = (float*)d_out;

    // workspace layout (all offsets power-of-two aligned), total ~62.3 MB
    char* ws = (char*)d_ws;
    unsigned short* fg    = (unsigned short*)(ws);              // 512*512*64 bf16 = 32 MB
    float*          norms = (float*)(ws + 33554432);            // 512*512 f32    =  1 MB
    unsigned short* qfb   = (unsigned short*)(ws + 34603008);   // 65536*64 bf16  =  8 MB
    float*          qnb   = (float*)(ws + 42991616);            // 65536 f32      = 256 KB
    float*          lbsT  = (float*)(ws + 43253760);            // 512*512*21 f32 = 22 MB

    upsample_kernel<<<kRC / 256, 256, 0, stream>>>(feat, fg, norms);
    query_kernel  <<<kM  / 256, 256, 0, stream>>>(feat, qfb, qnb);
    lbst_kernel   <<<kRC / 256, 256, 0, stream>>>(lbs, lbsT);
    vote_kernel   <<<kM / 8,    256, 0, stream>>>(grid_inte, grid, wei,
                                                  fg, norms, qfb, qnb, lbsT, out);
}
